// MoE_28157805592688
// MI455X (gfx1250) — matrix operand order repaired and checked by reading, not compiled, not run on hardware
//
#include <hip/hip_runtime.h>
#include <math.h>

// ---------------------------------------------------------------------------
// MoE layer for MI455X (gfx1250, wave32, WMMA).
//   att = (q@Wv+bv)@Wo+bo  (softmax over single score == 1, so Q/K are dead)
//   gate softmax / top-2 ; 8 dense experts with fused relu / comb-accumulate.
// Strategy: one-time fp32->fp16 conversion (weights also transposed to [N][K]),
// every GEMM runs on fp16 tiles staged to LDS with async b128 copies
// (double buffered, ASYNCcnt-tracked), computed with v_wmma_f32_16x16x32_f16.
// Block tile 128x128, wave tile 32x64 (2x4 WMMA tiles) for fragment reuse.
// ---------------------------------------------------------------------------

typedef _Float16 v4h  __attribute__((ext_vector_type(4)));
typedef _Float16 v8h  __attribute__((ext_vector_type(8)));
typedef _Float16 v16h __attribute__((ext_vector_type(16)));
typedef float    v8f  __attribute__((ext_vector_type(8)));
typedef int      v4i  __attribute__((ext_vector_type(4)));

#define DIM   512
#define NTOK  8192
#define NEXP  8

#if defined(__AMDGCN__) && __has_builtin(__builtin_amdgcn_global_load_async_to_lds_b128) && __has_builtin(__builtin_amdgcn_s_wait_asynccnt)
#  define HAVE_ASYNC_LDS 1
#else
#  define HAVE_ASYNC_LDS 0
#endif

#if HAVE_ASYNC_LDS
typedef __attribute__((address_space(1))) v4i GASv4i;   // global int4
typedef __attribute__((address_space(3))) v4i LASv4i;   // LDS int4
#endif

// Copy 16 bytes global(fp16) -> LDS(fp16).
__device__ __forceinline__ void copy16_g2l(const _Float16* g, _Float16* l)
{
#if HAVE_ASYNC_LDS
    __builtin_amdgcn_global_load_async_to_lds_b128(
        (GASv4i*)g, (LASv4i*)l, 0, 0);
#else
    *reinterpret_cast<v8h*>(l) = *reinterpret_cast<const v8h*>(g);
#endif
}

// ---------------------------------------------------------------------------
// GEMM: out = epilogue(A[M x 512](f16,row-major) @ Bt[512 x 512](f16, [N][K])
//                      + bias[512](f32))
//   mode 0: Ch = val            (fp16 out)
//   mode 1: Ch = relu(val)      (fp16 out)
//   mode 2: Cf (initOut ? = : +=) comb[row*8+expert] * val   (fp32 out)
// Block 128(M) x 128(N); 8 waves, each computing a 32x64 tile (2x4 WMMAs).
// ---------------------------------------------------------------------------
__global__ __launch_bounds__(256) void gemm512_wmma(
    const _Float16* __restrict__ A, const _Float16* __restrict__ Bt,
    const float* __restrict__ bias, void* __restrict__ Cout,
    const float* __restrict__ comb, int expert, int mode, int initOut)
{
    // Double-buffered LDS tiles: A [m][k] 128x32, B transposed [n][k] 128x32
    __shared__ __attribute__((aligned(32))) _Float16 sA[2][128 * 32];
    __shared__ __attribute__((aligned(32))) _Float16 sB[2][128 * 32];

    const int tid  = threadIdx.x;
    const int lane = tid & 31;
    const int wave = tid >> 5;     // 0..7
    const int lrow = lane & 15;
    const int hid  = lane >> 4;    // wave half (0/1)
    const int wm   = wave >> 1;    // 0..3 -> M offset wm*32
    const int wn   = wave & 1;     // 0..1 -> N offset wn*64
    const int m0   = blockIdx.y * 128;
    const int n0   = blockIdx.x * 128;

    // per-thread staging slots: 2 chunks of A + 2 chunks of B (16B each)
    const int r0 = tid >> 2,          k0o = (tid & 3) << 3;
    const int r1 = (tid + 256) >> 2;  // second chunk, same k offset

    auto copy_tile = [&](int kbase, int buf) {
        copy16_g2l(A  + (size_t)(m0 + r0) * 512 + kbase + k0o,
                   &sA[buf][r0 * 32 + k0o]);
        copy16_g2l(A  + (size_t)(m0 + r1) * 512 + kbase + k0o,
                   &sA[buf][r1 * 32 + k0o]);
        copy16_g2l(Bt + (size_t)(n0 + r0) * 512 + kbase + k0o,
                   &sB[buf][r0 * 32 + k0o]);
        copy16_g2l(Bt + (size_t)(n0 + r1) * 512 + kbase + k0o,
                   &sB[buf][r1 * 32 + k0o]);
    };

    v8f acc[2][4];
    #pragma unroll
    for (int mt = 0; mt < 2; ++mt)
        #pragma unroll
        for (int nt = 0; nt < 4; ++nt)
            acc[mt][nt] = {};

    copy_tile(0, 0);
    for (int t = 0; t < 16; ++t) {
        const int cur = t & 1;
        if (t + 1 < 16) copy_tile((t + 1) * 32, cur ^ 1);
#if HAVE_ASYNC_LDS
        // in-order completion: waiting <=4 guarantees tile t has landed while
        // tile t+1's 4 async copies stay in flight.
        if (t + 1 < 16) __builtin_amdgcn_s_wait_asynccnt(4);
        else            __builtin_amdgcn_s_wait_asynccnt(0);
#endif
        __syncthreads();

        // fragments per CDNA5 16-bit A/B VGPR layouts
        v16h af[2], bf[4];
        #pragma unroll
        for (int mt = 0; mt < 2; ++mt) {
            const int row = wm * 32 + mt * 16 + lrow;
            v8h lo = *reinterpret_cast<const v8h*>(&sA[cur][row * 32 + hid * 8]);
            v8h hi = *reinterpret_cast<const v8h*>(&sA[cur][row * 32 + 16 + hid * 8]);
            af[mt] = __builtin_shufflevector(lo, hi,
                0, 1, 2, 3, 4, 5, 6, 7, 8, 9, 10, 11, 12, 13, 14, 15);
        }
        #pragma unroll
        for (int nt = 0; nt < 4; ++nt) {
            const int col = wn * 64 + nt * 16 + lrow;
            v8h lo = *reinterpret_cast<const v8h*>(&sB[cur][col * 32 + hid * 8]);
            v8h hi = *reinterpret_cast<const v8h*>(&sB[cur][col * 32 + 16 + hid * 8]);
            bf[nt] = __builtin_shufflevector(lo, hi,
                0, 1, 2, 3, 4, 5, 6, 7, 8, 9, 10, 11, 12, 13, 14, 15);
        }

        #pragma unroll
        for (int mt = 0; mt < 2; ++mt)
            #pragma unroll
            for (int nt = 0; nt < 4; ++nt)
                acc[mt][nt] = __builtin_amdgcn_wmma_f32_16x16x32_f16(
                    false, af[mt], false, bf[nt],
                    (short)0, acc[mt][nt], false, false);
        __syncthreads();   // protect buf cur before next iter's copies overwrite it
    }

    // epilogue. C/D layout: VGPR r -> M = base + hid*8 + r ; N = lane%16
    _Float16* Ch = (_Float16*)Cout;
    float*    Cf = (float*)Cout;
    #pragma unroll
    for (int mt = 0; mt < 2; ++mt) {
        #pragma unroll
        for (int nt = 0; nt < 4; ++nt) {
            const int gcol = n0 + wn * 64 + nt * 16 + lrow;
            const float bv = bias[gcol];
            #pragma unroll
            for (int r = 0; r < 8; ++r) {
                const int grow = m0 + wm * 32 + mt * 16 + hid * 8 + r;
                const float val = acc[mt][nt][r] + bv;
                const size_t o = (size_t)grow * 512 + gcol;
                if (mode == 0) {
                    Ch[o] = (_Float16)val;
                } else if (mode == 1) {
                    Ch[o] = (_Float16)fmaxf(val, 0.0f);
                } else {
                    float res = comb[grow * NEXP + expert] * val;
                    if (!initOut) res += Cf[o];
                    Cf[o] = res;
                }
            }
        }
    }
}

// fp32 -> fp16, contiguous (n must be a multiple of 8; true for all uses)
__global__ __launch_bounds__(256) void cvt_f32_f16(
    const float* __restrict__ s, _Float16* __restrict__ d, int n)
{
    const int i = (blockIdx.x * 256 + threadIdx.x) * 8;
    if (i + 8 <= n) {
        const float4 a = *reinterpret_cast<const float4*>(&s[i]);
        const float4 b = *reinterpret_cast<const float4*>(&s[i + 4]);
        v8h h;
        h[0] = (_Float16)a.x; h[1] = (_Float16)a.y;
        h[2] = (_Float16)a.z; h[3] = (_Float16)a.w;
        h[4] = (_Float16)b.x; h[5] = (_Float16)b.y;
        h[6] = (_Float16)b.z; h[7] = (_Float16)b.w;
        *reinterpret_cast<v8h*>(&d[i]) = h;
    }
}

// Batched 512x512 fp32 [K][N] -> fp16 [N][K] (transpose + convert).
// grid (16,16,batch), 256 threads, 32x32 LDS tile.
__global__ __launch_bounds__(256) void cvt_t512(
    const float* __restrict__ src, _Float16* __restrict__ dst)
{
    __shared__ float tile[32][33];
    const float*  S = src + (size_t)blockIdx.z * DIM * DIM;
    _Float16*     D = dst + (size_t)blockIdx.z * DIM * DIM;
    const int k0 = blockIdx.y * 32, n0 = blockIdx.x * 32;

    const int ty = threadIdx.x >> 3;        // 0..31 (k-local)
    const int tx = threadIdx.x & 7;         // 8 float4 per row
    const float4 v = *reinterpret_cast<const float4*>(
        &S[(size_t)(k0 + ty) * DIM + n0 + tx * 4]);
    tile[ty][tx * 4 + 0] = v.x; tile[ty][tx * 4 + 1] = v.y;
    tile[ty][tx * 4 + 2] = v.z; tile[ty][tx * 4 + 3] = v.w;
    __syncthreads();

    const int oy = threadIdx.x >> 3;        // n-local
    const int ox = threadIdx.x & 7;         // k-chunk
    v4h h;
    #pragma unroll
    for (int j = 0; j < 4; ++j)
        h[j] = (_Float16)tile[ox * 4 + j][oy];
    *reinterpret_cast<v4h*>(&D[(size_t)(n0 + oy) * DIM + k0 + ox * 4]) = h;
}

// Gating over fp16 att: logits = att@gate_W + gate_b ; softmax ; top-2.
__global__ __launch_bounds__(256) void gate_kernel(
    const _Float16* __restrict__ att, const float* __restrict__ gW,
    const float* __restrict__ gb, float* __restrict__ gate_prob,
    float* __restrict__ comb, float* __restrict__ importance)
{
    const int row = blockIdx.x * blockDim.x + threadIdx.x;
    if (row >= NTOK) return;

    float lg[NEXP];
    #pragma unroll
    for (int j = 0; j < NEXP; ++j) lg[j] = gb[j];

    const _Float16* ar = &att[(size_t)row * DIM];
    for (int k = 0; k < DIM; ++k) {
        const float a = (float)ar[k];
        const float4 w0 = *reinterpret_cast<const float4*>(&gW[k * NEXP]);
        const float4 w1 = *reinterpret_cast<const float4*>(&gW[k * NEXP + 4]);
        lg[0] += a * w0.x; lg[1] += a * w0.y; lg[2] += a * w0.z; lg[3] += a * w0.w;
        lg[4] += a * w1.x; lg[5] += a * w1.y; lg[6] += a * w1.z; lg[7] += a * w1.w;
    }

    float m = lg[0];
    #pragma unroll
    for (int j = 1; j < NEXP; ++j) m = fmaxf(m, lg[j]);
    float p[NEXP], s = 0.0f;
    #pragma unroll
    for (int j = 0; j < NEXP; ++j) { p[j] = __expf(lg[j] - m); s += p[j]; }
    const float inv = 1.0f / s;
    #pragma unroll
    for (int j = 0; j < NEXP; ++j) {
        p[j] *= inv;
        gate_prob[(size_t)row * NEXP + j] = p[j];
        atomicAdd(&importance[j], p[j]);
    }

    int i1 = 0;
    #pragma unroll
    for (int j = 1; j < NEXP; ++j) if (p[j] > p[i1]) i1 = j;
    int i2 = (i1 == 0) ? 1 : 0;
    #pragma unroll
    for (int j = 0; j < NEXP; ++j) if (j != i1 && p[j] > p[i2]) i2 = j;

    const float e2 = __expf(p[i2] - p[i1]);
    const float z  = 1.0f + e2;
    float cw[NEXP];
    #pragma unroll
    for (int j = 0; j < NEXP; ++j) cw[j] = 0.0f;
    cw[i1] = 1.0f / z;
    cw[i2] = e2 / z;
    #pragma unroll
    for (int j = 0; j < NEXP; ++j) comb[(size_t)row * NEXP + j] = cw[j];
}

__global__ void zero_kernel(float* __restrict__ p, int n)
{
    const int i = blockIdx.x * blockDim.x + threadIdx.x;
    if (i < n) p[i] = 0.0f;
}

__global__ void loss_kernel(const float* __restrict__ importance,
                            float* __restrict__ out)
{
    if (threadIdx.x == 0) {
        float s = 0.0f;
        for (int j = 0; j < NEXP; ++j) s += importance[j];
        const float mean = s / (float)NEXP;
        float var = 0.0f;
        for (int j = 0; j < NEXP; ++j) {
            const float d = importance[j] - mean;
            var += d * d;
        }
        var /= (float)(NEXP - 1);          // ddof=1
        const float cv = sqrtf(var) / mean;
        out[0] = 0.01f * cv * cv;          // W_IMP
    }
}

extern "C" void kernel_launch(void* const* d_in, const int* in_sizes, int n_in,
                              void* d_out, int out_size, void* d_ws, size_t ws_size,
                              hipStream_t stream)
{
    (void)in_sizes; (void)n_in; (void)out_size; (void)ws_size;

    const float* x  = (const float*)d_in[0];
    const float* q  = (const float*)d_in[1];
    const float* Wv = (const float*)d_in[6];
    const float* bv = (const float*)d_in[7];
    const float* Wo = (const float*)d_in[8];
    const float* bo = (const float*)d_in[9];
    const float* gW = (const float*)d_in[10];
    const float* gb = (const float*)d_in[11];
    const float* W1 = (const float*)d_in[12];
    const float* b1 = (const float*)d_in[13];
    const float* W2 = (const float*)d_in[14];
    const float* b2 = (const float*)d_in[15];
    // d_in[2..5] (Wq,bq,Wk,bk) are mathematically dead; d_in[16] top == 2.

    // workspace: fp16 region first, then fp32 comb/importance
    _Float16* hp  = (_Float16*)d_ws;
    _Float16* hq  = hp; hp += (size_t)NTOK * DIM;
    _Float16* hx  = hp; hp += (size_t)NTOK * DIM;
    _Float16* hWv = hp; hp += (size_t)DIM * DIM;
    _Float16* hWo = hp; hp += (size_t)DIM * DIM;
    _Float16* hW1 = hp; hp += (size_t)NEXP * DIM * DIM;
    _Float16* hW2 = hp; hp += (size_t)NEXP * DIM * DIM;
    _Float16* Vh  = hp; hp += (size_t)NTOK * DIM;   // V, reused as expert h
    _Float16* atH = hp; hp += (size_t)NTOK * DIM;
    float* comb = (float*)hp;
    float* imp  = comb + (size_t)NTOK * NEXP;

    float* y         = (float*)d_out;
    float* gate_prob = y + (size_t)NTOK * DIM;
    float* loss      = gate_prob + (size_t)NTOK * NEXP;

    const dim3 gblk(256);
    const dim3 ggrid(DIM / 128, NTOK / 128);             // (4, 64)
    const int  cvtBlocks = (NTOK * DIM) / (256 * 8);     // 2048

    zero_kernel<<<dim3(1), dim3(32), 0, stream>>>(imp, NEXP);

    // one-time precision/layout conversion
    cvt_f32_f16<<<dim3(cvtBlocks), gblk, 0, stream>>>(q, hq, NTOK * DIM);
    cvt_f32_f16<<<dim3(cvtBlocks), gblk, 0, stream>>>(x, hx, NTOK * DIM);
    cvt_t512<<<dim3(16, 16, 1), gblk, 0, stream>>>(Wv, hWv);
    cvt_t512<<<dim3(16, 16, 1), gblk, 0, stream>>>(Wo, hWo);
    cvt_t512<<<dim3(16, 16, NEXP), gblk, 0, stream>>>(W1, hW1);
    cvt_t512<<<dim3(16, 16, NEXP), gblk, 0, stream>>>(W2, hW2);

    // att = (q@Wv + bv) @ Wo + bo
    gemm512_wmma<<<ggrid, gblk, 0, stream>>>(hq, hWv, bv, Vh,  nullptr, 0, 0, 0);
    gemm512_wmma<<<ggrid, gblk, 0, stream>>>(Vh, hWo, bo, atH, nullptr, 0, 0, 0);

    gate_kernel<<<dim3(NTOK / 256), gblk, 0, stream>>>(atH, gW, gb,
                                                       gate_prob, comb, imp);

    for (int e = 0; e < NEXP; ++e) {
        gemm512_wmma<<<ggrid, gblk, 0, stream>>>(
            hx, hW1 + (size_t)e * DIM * DIM, b1 + (size_t)e * DIM,
            Vh, nullptr, 0, 1, 0);                         // h = relu(xf@W1+b1)
        gemm512_wmma<<<ggrid, gblk, 0, stream>>>(
            Vh, hW2 + (size_t)e * DIM * DIM, b2 + (size_t)e * DIM,
            y, comb, e, 2, (e == 0) ? 1 : 0);              // y (+)= comb*(h@W2+b2)
    }

    loss_kernel<<<dim3(1), dim3(32), 0, stream>>>(imp, loss);
}
